// GATBlock_42047729828495
// MI455X (gfx1250) — matrix operand order repaired and checked by reading, not compiled, not run on hardware
//
#include <hip/hip_runtime.h>
#include <hip/hip_bf16.h>

typedef _Float16 v16h __attribute__((ext_vector_type(16)));
typedef float    v8f  __attribute__((ext_vector_type(8)));

#define F_IN  128
#define F_OUT 64
#define NEG_SLOPE 0.2f

// ---------------------------------------------------------------------------
// K1: h = x @ W via V_WMMA_F32_16X16X32_F16. One wave per 16x16 output tile,
// K=128 fully unrolled into 4 back-to-back WMMAs.
// ---------------------------------------------------------------------------
__global__ __launch_bounds__(256) void k_gemm_wmma(const float* __restrict__ x,
                                                   const float* __restrict__ Wm,
                                                   float* __restrict__ h,
                                                   int nNodes) {
    const int wave = threadIdx.x >> 5;
    const int lane = threadIdx.x & 31;
    const int nTilesN = F_OUT / 16;                       // 4
    const int nTilesM = (nNodes + 15) >> 4;               // 3125
    const int tile = blockIdx.x * 8 + wave;               // wave-uniform
    if (tile >= nTilesM * nTilesN) return;                // uniform exit per wave

    const int m0   = (tile / nTilesN) * 16;
    const int n0   = (tile % nTilesN) * 16;
    const int half = lane >> 4;                           // 0: lanes 0-15, 1: 16-31
    const int lr   = lane & 15;

    const int arow = min(m0 + lr, nNodes - 1);

    v8f c = {};
#pragma unroll
    for (int kb = 0; kb < F_IN; kb += 32) {
        v16h a, b;
        const float* xrow = x + (size_t)arow * F_IN + kb;
        // A-matrix 16x32 f16 layout (ISA 7.12.2): VGPR v<4 -> K = half*8 + 2v + p,
        // VGPR v>=4 -> K = 16 + half*8 + 2(v-4) + p. Pairs are K-contiguous -> float2.
#pragma unroll
        for (int v = 0; v < 8; ++v) {
            const int kA = (v < 4) ? ((half ? 8 : 0) + 2 * v)
                                   : (16 + (half ? 8 : 0) + 2 * (v - 4));
            const float2 f = *(const float2*)(xrow + kA);
            a[2 * v]     = (_Float16)f.x;
            a[2 * v + 1] = (_Float16)f.y;
        }
        // B-matrix 32x16: same K order as A: e<8 -> K = half*8 + e, e>=8 -> K = 16 + half*8 + (e-8)
#pragma unroll
        for (int e = 0; e < 16; ++e) {
            b[e] = (_Float16)Wm[(size_t)(kb + half * 8 + e + (e >= 8 ? 8 : 0)) * F_OUT + (n0 + lr)];
        }
        c = __builtin_amdgcn_wmma_f32_16x16x32_f16(false, a, false, b,
                                                   (short)0, c, false, false);
    }

    // C/D layout: VGPR r -> row m0 + half*8 + r, col n0 + lr
    float* hout = h + (size_t)(m0 + half * 8) * F_OUT + (n0 + lr);
    if (m0 + 16 <= nNodes) {                              // uniform fast path
#pragma unroll
        for (int r = 0; r < 8; ++r)
            hout[(size_t)r * F_OUT] = c[r];
    } else {                                              // ragged final tile
        for (int r = 0; r < 8; ++r)
            if (m0 + half * 8 + r < nNodes)
                hout[(size_t)r * F_OUT] = c[r];
    }
}

// ---------------------------------------------------------------------------
// K2: per-node attention logits a_src[n] = h[n]·att_src, a_dst[n] = h[n]·att_dst
// ---------------------------------------------------------------------------
__global__ void k_node_logits(const float4* __restrict__ h4,
                              const float* __restrict__ att_s,
                              const float* __restrict__ att_d,
                              float* __restrict__ a_s,
                              float* __restrict__ a_d, int n) {
    const int i = blockIdx.x * blockDim.x + threadIdx.x;
    if (i >= n) return;
    const float4* row = h4 + (size_t)i * (F_OUT / 4);
    float s = 0.f, d = 0.f;
#pragma unroll
    for (int j = 0; j < F_OUT / 4; ++j) {
        const float4 v = row[j];
        s += v.x * att_s[4*j] + v.y * att_s[4*j+1] + v.z * att_s[4*j+2] + v.w * att_s[4*j+3];
        d += v.x * att_d[4*j] + v.y * att_d[4*j+1] + v.z * att_d[4*j+2] + v.w * att_d[4*j+3];
    }
    a_s[i] = s;
    a_d[i] = d;
}

// order-preserving float <-> uint encoding for atomic segment-max
__device__ __forceinline__ unsigned enc_f32(float f) {
    unsigned u = __float_as_uint(f);
    return (u & 0x80000000u) ? ~u : (u | 0x80000000u);
}
__device__ __forceinline__ float dec_f32(unsigned u) {
    return __uint_as_float((u & 0x80000000u) ? (u ^ 0x80000000u) : ~u);
}

// ---------------------------------------------------------------------------
// K3: e = leakyrelu(a_src[src] + a_dst[dst]); segment-max over dst
// edges [0,E) from edge_index, [E, E+N) are self loops
// ---------------------------------------------------------------------------
__global__ void k_edge_logits(const int* __restrict__ ei,
                              const float* __restrict__ a_s,
                              const float* __restrict__ a_d,
                              float* __restrict__ ebuf,
                              unsigned* __restrict__ emax,
                              int E_, int EA) {
    const int i = blockIdx.x * blockDim.x + threadIdx.x;
    if (i >= EA) return;
    const int s = (i < E_) ? ei[i]      : (i - E_);
    const int d = (i < E_) ? ei[E_ + i] : (i - E_);
    float e = a_s[s] + a_d[d];
    e = (e > 0.f) ? e : NEG_SLOPE * e;
    ebuf[i] = e;
    atomicMax(&emax[d], enc_f32(e));
}

// ---------------------------------------------------------------------------
// K4: e_exp = exp(e - emax[dst]); segment-sum over dst
// ---------------------------------------------------------------------------
__global__ void k_edge_exp(const int* __restrict__ ei,
                           float* __restrict__ ebuf,
                           const unsigned* __restrict__ emax,
                           float* __restrict__ esum,
                           int E_, int EA) {
    const int i = blockIdx.x * blockDim.x + threadIdx.x;
    if (i >= EA) return;
    const int d = (i < E_) ? ei[E_ + i] : (i - E_);
    const float ex = __expf(ebuf[i] - dec_f32(emax[d]));
    ebuf[i] = ex;
    atomicAdd(&esum[d], ex);
}

// ---------------------------------------------------------------------------
// K5: out[dst] += (e_exp/e_sum[dst]) * h[src]; 64 threads per edge
// ---------------------------------------------------------------------------
__global__ void k_edge_scatter(const int* __restrict__ ei,
                               const float* __restrict__ ebuf,
                               const float* __restrict__ esum,
                               const float* __restrict__ h,
                               float* __restrict__ out,
                               int E_, int EA) {
    const long long t = (long long)blockIdx.x * blockDim.x + threadIdx.x;
    const int i = (int)(t >> 6);
    const int f = (int)(t & 63);
    if (i >= EA) return;
    const int s = (i < E_) ? ei[i]      : (i - E_);
    const int d = (i < E_) ? ei[E_ + i] : (i - E_);
    const float alpha = ebuf[i] / esum[d];
    atomicAdd(&out[(size_t)d * F_OUT + f], h[(size_t)s * F_OUT + f] * alpha);
}

// ---------------------------------------------------------------------------
// K6: out = relu(out + bias)
// ---------------------------------------------------------------------------
__global__ void k_finalize(float* __restrict__ out,
                           const float* __restrict__ bias, int total) {
    const int i = blockIdx.x * blockDim.x + threadIdx.x;
    if (i >= total) return;
    const float v = out[i] + bias[i & (F_OUT - 1)];
    out[i] = fmaxf(v, 0.f);
}

extern "C" void kernel_launch(void* const* d_in, const int* in_sizes, int n_in,
                              void* d_out, int out_size, void* d_ws, size_t ws_size,
                              hipStream_t stream) {
    const float* x     = (const float*)d_in[0];
    const int*   ei    = (const int*)d_in[1];
    const float* Wm    = (const float*)d_in[2];
    const float* att_s = (const float*)d_in[3];
    const float* att_d = (const float*)d_in[4];
    const float* bias  = (const float*)d_in[5];
    float* out = (float*)d_out;

    const int N  = in_sizes[0] / F_IN;   // 50000
    const int E  = in_sizes[1] / 2;      // 1600000
    const int EA = E + N;                // edges + self loops

    // workspace layout
    float*    h    = (float*)d_ws;              // N*64
    float*    a_s  = h + (size_t)N * F_OUT;     // N
    float*    a_d  = a_s + N;                   // N
    unsigned* emax = (unsigned*)(a_d + N);      // N
    float*    esum = (float*)(emax + N);        // N
    float*    ebuf = esum + N;                  // EA

    hipMemsetAsync(emax, 0, (size_t)N * sizeof(unsigned), stream);  // enc(-inf-ish)
    hipMemsetAsync(esum, 0, (size_t)N * sizeof(float), stream);
    hipMemsetAsync(out,  0, (size_t)N * F_OUT * sizeof(float), stream);

    // K1: GEMM via WMMA. 16x16 tiles: (N/16)*(64/16) tiles, 8 waves/block.
    {
        const int tiles = ((N + 15) / 16) * (F_OUT / 16);
        const int blocks = (tiles + 7) / 8;
        k_gemm_wmma<<<blocks, 256, 0, stream>>>(x, Wm, h, N);
    }
    // K2: node logits
    k_node_logits<<<(N + 255) / 256, 256, 0, stream>>>((const float4*)h, att_s, att_d, a_s, a_d, N);
    // K3: edge logits + segment max
    k_edge_logits<<<(EA + 255) / 256, 256, 0, stream>>>(ei, a_s, a_d, ebuf, emax, E, EA);
    // K4: exp + segment sum
    k_edge_exp<<<(EA + 255) / 256, 256, 0, stream>>>(ei, ebuf, emax, esum, E, EA);
    // K5: weighted scatter aggregate (64 threads per edge)
    {
        const long long tot = (long long)EA * F_OUT;
        const int blocks = (int)((tot + 255) / 256);
        k_edge_scatter<<<blocks, 256, 0, stream>>>(ei, ebuf, esum, h, out, E, EA);
    }
    // K6: bias + relu
    k_finalize<<<(N * F_OUT + 255) / 256, 256, 0, stream>>>(out, bias, N * F_OUT);
}
